// Attention_40510131536381
// MI455X (gfx1250) — matrix operand order repaired and checked by reading, not compiled, not run on hardware
//
#include <hip/hip_runtime.h>

// ---------------------------------------------------------------------------
// out = softmax(mask(Q K^T)) V,  Q,K,V = split(x @ W_qkv)
// B=2, S=2048, DIM=1024, HEADS=16, HEAD_DIM=64.
// Matrix math: v_wmma_f32_16x16x32_bf16 (bf16 in, f32 accumulate).
// Pipeline: cvt_x, cvt_wt (transpose), qkv_gemm (WMMA), flash_attn
// (WMMA + online softmax, K/V tiles staged in LDS via the Tensor Data Mover).
// ---------------------------------------------------------------------------

#define BATCH   2
#define SEQ     2048
#define DIM     1024
#define N3      3072
#define HEADS   16
#define HD      64
#define NEGBIG  (-1e9f)

typedef __attribute__((ext_vector_type(16))) __bf16 v16bf;
typedef __attribute__((ext_vector_type(8)))  __bf16 v8bf;
typedef __attribute__((ext_vector_type(4)))  __bf16 v4bf;
typedef __attribute__((ext_vector_type(4)))  float  v4f;
typedef __attribute__((ext_vector_type(8)))  float  v8f;
typedef __attribute__((ext_vector_type(4)))  unsigned int u32x4;
typedef __attribute__((ext_vector_type(4)))  int    i32x4;
typedef __attribute__((ext_vector_type(8)))  int    i32x8;

union Frag16 { v16bf v; v8bf h[2]; };

// ---------------------------------------------------------------------------
// LDS offset of a __shared__ object (generic -> AS(3) -> 32-bit offset).
// Guarded: the addrspacecast is only valid in the device pass.
// ---------------------------------------------------------------------------
__device__ __forceinline__ unsigned lds_off(const void* p) {
#if defined(__HIP_DEVICE_COMPILE__)
    return (unsigned)(unsigned long long)
           (const __attribute__((address_space(3))) void*)p;
#else
    (void)p;
    return 0;
#endif
}

// ---------------------------------------------------------------------------
// TDM: async 2-D tile load global -> LDS.  rows x rowlen bf16 elements,
// row stride (elements).  D# per CDNA5 ISA ch.8 (data_size=1 -> 2 bytes).
// ---------------------------------------------------------------------------
__device__ __forceinline__ void tdm_load_2d(const void* lds_dst, const void* gsrc,
                                            unsigned rowlen, unsigned rows,
                                            unsigned long long stride_elems) {
    const unsigned long long ga = (unsigned long long)gsrc;

    u32x4 g0;
    g0[0] = 1u;                                        // count=1, user D#
    g0[1] = lds_off(lds_dst);                          // lds_addr (bytes)
    g0[2] = (unsigned)(ga & 0xFFFFFFFFu);              // global_addr[31:0]
    g0[3] = (unsigned)((ga >> 32) & 0x01FFFFFFu)       // global_addr[56:32]
          | (2u << 30);                                // type=2 ("image")

    i32x8 g1;
    g1[0] = (int)(1u << 16);                           // data_size=1 (2B), no mcast
    g1[1] = (int)((rowlen & 0xFFFFu) << 16);           // tensor_dim0[15:0]
    g1[2] = (int)(((rowlen >> 16) & 0xFFFFu)           // tensor_dim0[31:16]
          |  ((rows & 0xFFFFu) << 16));                // tensor_dim1[15:0]
    g1[3] = (int)(((rows >> 16) & 0xFFFFu)             // tensor_dim1[31:16]
          |  ((rowlen & 0xFFFFu) << 16));              // tile_dim0
    g1[4] = (int)(rows & 0xFFFFu);                     // tile_dim1 (tile_dim2=0)
    g1[5] = (int)(stride_elems & 0xFFFFFFFFu);         // tensor_dim0_stride[31:0]
    g1[6] = (int)((stride_elems >> 32) & 0xFFFFu);     // [47:32]; dim1_stride=0
    g1[7] = 0;

    i32x4 gz = {0, 0, 0, 0};                           // groups 2/3 unused (2-D)
#if __clang_major__ >= 23
    i32x8 gz8 = {0, 0, 0, 0, 0, 0, 0, 0};
    __builtin_amdgcn_tensor_load_to_lds(g0, g1, gz, gz, gz8, 0);
#else
    __builtin_amdgcn_tensor_load_to_lds(g0, g1, gz, gz, 0);
#endif
}

// ---------------------------------------------------------------------------
// x (f32 [4096,1024]) -> bf16, flat, vectorized x4
// ---------------------------------------------------------------------------
__global__ __launch_bounds__(256)
void cvt_x(const v4f* __restrict__ in, v4bf* __restrict__ out) {
    const int i = blockIdx.x * 256 + threadIdx.x;
    out[i] = __builtin_convertvector(in[i], v4bf);
}

// ---------------------------------------------------------------------------
// W (f32 [1024,3072]) -> Wt (bf16 [3072,1024]), 32x32 LDS-tiled transpose
// ---------------------------------------------------------------------------
__global__ __launch_bounds__(256)
void cvt_wt(const float* __restrict__ w, __bf16* __restrict__ wt) {
    __shared__ __bf16 tile[32][33];
    const int tx = threadIdx.x & 31;
    const int ty = threadIdx.x >> 5;
    const int n0 = blockIdx.x * 32;
    const int k0 = blockIdx.y * 32;
#pragma unroll
    for (int r = ty; r < 32; r += 8)
        tile[r][tx] = (__bf16)w[(size_t)(k0 + r) * N3 + n0 + tx];
    __syncthreads();
#pragma unroll
    for (int r = ty; r < 32; r += 8)
        wt[(size_t)(n0 + r) * DIM + k0 + tx] = tile[tx][r];
}

// ---------------------------------------------------------------------------
// qkv = xb @ Wt^T  (M=4096, K=1024, N=3072), all-bf16 contiguous loads.
// One wave computes 32x64: 8 WMMAs per K-step of 32.
// Q,K head-major [B][H][S][64]; V stored transposed [B][H][64][S].
// ---------------------------------------------------------------------------
__global__ __launch_bounds__(256)
void qkv_gemm(const __bf16* __restrict__ xb, const __bf16* __restrict__ wt,
              __bf16* __restrict__ Q, __bf16* __restrict__ K,
              __bf16* __restrict__ Vt) {
    const int lane = threadIdx.x & 31;
    const int l16  = lane & 15;
    const int hi   = lane >> 4;
    const int wid  = blockIdx.x * (blockDim.x >> 5) + (threadIdx.x >> 5);
    const int nW   = N3 / 64;
    const int m0   = (wid / nW) * 32;
    const int n0   = (wid % nW) * 64;

    v8f acc[2][4];
#pragma unroll
    for (int mi = 0; mi < 2; ++mi)
#pragma unroll
        for (int t = 0; t < 4; ++t)
#pragma unroll
            for (int j = 0; j < 8; ++j) acc[mi][t][j] = 0.0f;

    const __bf16* xr0 = xb + (size_t)(m0 + l16) * DIM + hi * 8;
    const __bf16* xr1 = xr0 + (size_t)16 * DIM;
    const __bf16* wr0 = wt + (size_t)(n0 + l16) * DIM + hi * 8;

    for (int kc = 0; kc < DIM; kc += 32) {
        Frag16 a[2];
        a[0].h[0] = *(const v8bf*)(xr0 + kc);
        a[0].h[1] = *(const v8bf*)(xr0 + kc + 16);
        a[1].h[0] = *(const v8bf*)(xr1 + kc);
        a[1].h[1] = *(const v8bf*)(xr1 + kc + 16);
#pragma unroll
        for (int t = 0; t < 4; ++t) {
            Frag16 bfv;
            const __bf16* wr = wr0 + (size_t)t * 16 * DIM + kc;
            bfv.h[0] = *(const v8bf*)(wr);
            bfv.h[1] = *(const v8bf*)(wr + 16);
            acc[0][t] = __builtin_amdgcn_wmma_f32_16x16x32_bf16(
                false, a[0].v, false, bfv.v, (short)0, acc[0][t], false, false);
            acc[1][t] = __builtin_amdgcn_wmma_f32_16x16x32_bf16(
                false, a[1].v, false, bfv.v, (short)0, acc[1][t], false, false);
        }
    }

#pragma unroll
    for (int mi = 0; mi < 2; ++mi) {
        const int rowb = m0 + mi * 16 + hi * 8;
#pragma unroll
        for (int t = 0; t < 4; ++t) {
            const int colbase = n0 + t * 16;
            const int which   = colbase / DIM;        // wave-uniform
            const int dimb    = colbase % DIM;
            const int h       = dimb / HD;
            const int dd      = (dimb % HD) + l16;
            if (which == 2) {
#pragma unroll
                for (int j = 0; j < 8; ++j) {
                    const int row = rowb + j;
                    const int b   = row >> 11;
                    const int s   = row & (SEQ - 1);
                    Vt[(((size_t)b * HEADS + h) * HD + dd) * SEQ + s] =
                        (__bf16)acc[mi][t][j];
                }
            } else {
                __bf16* dst = (which == 0) ? Q : K;
#pragma unroll
                for (int j = 0; j < 8; ++j) {
                    const int row = rowb + j;
                    const int b   = row >> 11;
                    const int s   = row & (SEQ - 1);
                    dst[(((size_t)b * HEADS + h) * SEQ + s) * HD + dd] =
                        (__bf16)acc[mi][t][j];
                }
            }
        }
    }
}

// ---------------------------------------------------------------------------
// Flash attention.  Block = 4 waves x 16 query rows = 64 queries.
// Per 32-key tile: TDM stages K (32x64) and V^T (64x32) into LDS once per
// block, then 4 WMMAs (QK^T) + online softmax + 4 WMMAs (PV).
// ---------------------------------------------------------------------------
__global__ __launch_bounds__(128)
void flash_attn(const __bf16* __restrict__ Q, const __bf16* __restrict__ K,
                const __bf16* __restrict__ Vt, const int* __restrict__ mask,
                float* __restrict__ out) {
    const int lane = threadIdx.x & 31;
    const int l16  = lane & 15;
    const int hi   = lane >> 4;
    const int wv   = threadIdx.x >> 5;
    const int pair = blockIdx.x >> 5;                 // b*16+h
    const int qblk = blockIdx.x & 31;
    const int b    = pair >> 4;
    const int h    = pair & 15;
    const int q0   = qblk * 64 + wv * 16;

    const __bf16* Qbh  = Q  + ((size_t)pair * SEQ) * HD;
    const __bf16* Kbh  = K  + ((size_t)pair * SEQ) * HD;
    const __bf16* Vtbh = Vt + ((size_t)pair * HD) * SEQ;
    const int*    mb   = mask + b * SEQ;

    __shared__ alignas(128) __bf16 ldsK[32 * HD];      // keys tile: [32][64]
    __shared__ alignas(128) __bf16 ldsV[HD * 32];      // V^T tile:  [64][32]
    __shared__ alignas(64)  __bf16 pst[4][16][32];     // per-wave P staging

    // Q A-frags for K-dim 0..31 and 32..63 (register-resident)
    Frag16 aQ[2];
    {
        const __bf16* qr = Qbh + (size_t)(q0 + l16) * HD + hi * 8;
#pragma unroll
        for (int c = 0; c < 2; ++c) {
            aQ[c].h[0] = *(const v8bf*)(qr + c * 32);
            aQ[c].h[1] = *(const v8bf*)(qr + c * 32 + 16);
        }
    }

    v8f acc[4];
    float mrow[8], lrow[8];
#pragma unroll
    for (int t = 0; t < 4; ++t)
#pragma unroll
        for (int j = 0; j < 8; ++j) acc[t][j] = 0.0f;
#pragma unroll
    for (int j = 0; j < 8; ++j) { mrow[j] = -INFINITY; lrow[j] = 0.0f; }

    for (int k0 = 0; k0 < SEQ; k0 += 32) {
        __syncthreads();                               // tile consumed
        if (wv == 0) {
            // K tile: 32 key-rows x 64 dims, row stride 64
            tdm_load_2d(ldsK, Kbh + (size_t)k0 * HD, HD, 32, HD);
            // V^T tile: 64 dim-rows x 32 keys, row stride SEQ
            tdm_load_2d(ldsV, Vtbh + k0, 32, HD, SEQ);
            __builtin_amdgcn_s_wait_tensorcnt(0);
        }
        __syncthreads();                               // tile published

        // S = Q K^T: 16x32 score tile (two 16x16 C tiles)
        v8f sc[2];
#pragma unroll
        for (int nt = 0; nt < 2; ++nt) {
#pragma unroll
            for (int j = 0; j < 8; ++j) sc[nt][j] = 0.0f;
#pragma unroll
            for (int c = 0; c < 2; ++c) {
                Frag16 bK;
                const __bf16* kr = ldsK + (nt * 16 + l16) * HD + c * 32 + hi * 8;
                bK.h[0] = *(const v8bf*)(kr);
                bK.h[1] = *(const v8bf*)(kr + 16);
                sc[nt] = __builtin_amdgcn_wmma_f32_16x16x32_bf16(
                    false, aQ[c].v, false, bK.v, (short)0, sc[nt], false, false);
            }
        }

        // mask + online softmax (row = j + 8*hi, cols across the 16-lane half)
        const int mk0 = mb[k0 + l16];
        const int mk1 = mb[k0 + 16 + l16];
        float s0[8], s1[8], tm[8];
#pragma unroll
        for (int j = 0; j < 8; ++j) {
            s0[j] = mk0 ? sc[0][j] : NEGBIG;
            s1[j] = mk1 ? sc[1][j] : NEGBIG;
            tm[j] = fmaxf(s0[j], s1[j]);
        }
#pragma unroll
        for (int o = 1; o < 16; o <<= 1)
#pragma unroll
            for (int j = 0; j < 8; ++j)
                tm[j] = fmaxf(tm[j], __shfl_xor(tm[j], o, 32));

        float p0[8], p1[8], rs[8];
#pragma unroll
        for (int j = 0; j < 8; ++j) {
            const float mn  = fmaxf(mrow[j], tm[j]);
            const float scl = __expf(mrow[j] - mn);
            p0[j] = __expf(s0[j] - mn);
            p1[j] = __expf(s1[j] - mn);
            rs[j] = p0[j] + p1[j];
            mrow[j] = mn;
            lrow[j] *= scl;
#pragma unroll
            for (int t = 0; t < 4; ++t) acc[t][j] *= scl;
        }
#pragma unroll
        for (int o = 1; o < 16; o <<= 1)
#pragma unroll
            for (int j = 0; j < 8; ++j) rs[j] += __shfl_xor(rs[j], o, 32);
#pragma unroll
        for (int j = 0; j < 8; ++j) lrow[j] += rs[j];

        // C-layout f32 -> A-layout bf16 via per-wave LDS staging
#pragma unroll
        for (int j = 0; j < 8; ++j) {
            const int r = j + hi * 8;
            pst[wv][r][l16]      = (__bf16)p0[j];
            pst[wv][r][16 + l16] = (__bf16)p1[j];
        }
        Frag16 aP;
        aP.h[0] = *(const v8bf*)&pst[wv][l16][hi * 8];
        aP.h[1] = *(const v8bf*)&pst[wv][l16][16 + hi * 8];

        // O += P @ V: B(kappa,n) = ldsV[n][kappa], table order per lane
#pragma unroll
        for (int t = 0; t < 4; ++t) {
            Frag16 bV;
            const __bf16* vr = ldsV + (t * 16 + l16) * 32 + hi * 8;
            bV.h[0] = *(const v8bf*)(vr);
            bV.h[1] = *(const v8bf*)(vr + 16);
            acc[t] = __builtin_amdgcn_wmma_f32_16x16x32_bf16(
                false, aP.v, false, bV.v, (short)0, acc[t], false, false);
        }
    }

    // Epilogue: divide by running sum, write out[b][s][h*64+d]
    float linv[8];
#pragma unroll
    for (int j = 0; j < 8; ++j) linv[j] = 1.0f / lrow[j];
#pragma unroll
    for (int t = 0; t < 4; ++t) {
        const int d = h * HD + t * 16 + l16;
#pragma unroll
        for (int j = 0; j < 8; ++j) {
            const int s = q0 + j + hi * 8;
            out[((size_t)b * SEQ + s) * DIM + d] = acc[t][j] * linv[j];
        }
    }
}

// ---------------------------------------------------------------------------
extern "C" void kernel_launch(void* const* d_in, const int* in_sizes, int n_in,
                              void* d_out, int out_size, void* d_ws, size_t ws_size,
                              hipStream_t stream) {
    const float* x    = (const float*)d_in[0];   // [2, 2048, 1024]
    const float* w    = (const float*)d_in[1];   // [1024, 3072]
    const int*   mask = (const int*)d_in[2];     // [2, 2048]
    float*       out  = (float*)d_out;           // [2, 2048, 1024]

    const size_t nX   = (size_t)BATCH * SEQ * DIM;        // 4M
    const size_t nW   = (size_t)DIM * N3;                 // 3M
    const size_t nQKV = (size_t)BATCH * HEADS * SEQ * HD; // 4M

    __bf16* xb  = (__bf16*)d_ws;          // 8 MB
    __bf16* wtb = xb  + nX;               // 6 MB
    __bf16* Qws = wtb + nW;               // 8 MB
    __bf16* Kws = Qws + nQKV;             // 8 MB
    __bf16* Vts = Kws + nQKV;             // 8 MB (transposed [B][H][64][S])

    cvt_x<<<(int)(nX / 4 / 256), 256, 0, stream>>>((const v4f*)x, (v4bf*)xb);
    cvt_wt<<<dim3(N3 / 32, DIM / 32), 256, 0, stream>>>(w, wtb);
    qkv_gemm<<<768, 256, 0, stream>>>(xb, wtb, Qws, Kws, Vts);
    flash_attn<<<1024, 128, 0, stream>>>(Qws, Kws, Vts, mask, out);
}
